// AriaTextMoELayer_24172075942097
// MI455X (gfx1250) — matrix operand order repaired and checked by reading, not compiled, not run on hardware
//
#include <hip/hip_runtime.h>

typedef __bf16 bf16;
typedef __bf16 v16bf __attribute__((ext_vector_type(16)));
typedef float  v8f   __attribute__((ext_vector_type(8)));

#define E_    16
#define TOPK_ 4
#define D_    2560
#define F_    1664
#define FS_   3328
#define T_    2048

// GEMM tiling
#define BM 128
#define BN 64
#define BK 32

// ---------------------------------------------------------------------------
// WMMA helper (wave32, v_wmma_f32_16x16x32_bf16)
// ---------------------------------------------------------------------------
__device__ __forceinline__ v8f wmma_bf16(v16bf a, v16bf b, v8f c) {
    return __builtin_amdgcn_wmma_f32_16x16x32_bf16(
        /*neg_a=*/false, a, /*neg_b=*/false, b,
        /*c_mod=*/(short)0, c, /*reuse_a=*/false, /*reuse_b=*/false);
}

// Fragment load from a pre-swizzled LDS buffer organized as
// [tile16][32 lane slots][16 bf16] -> two ds_load_b128 per fragment.
__device__ __forceinline__ v16bf frag_ld(const bf16* buf, int tile16) {
    int lane = threadIdx.x & 31;
    const uint4* p = (const uint4*)(buf + ((tile16 * 32 + lane) << 4));
    union { uint4 u[2]; v16bf v; } t;
    t.u[0] = p[0];
    t.u[1] = p[1];
    return t.v;
}

// Swizzled A staging: thread holds 16 contiguous bf16 of row r, k-half `half`
// (half=0 -> K 0..15, half=1 -> K 16..31).
//   A frag layout: lane l<16: row=l, K {0..7, 16..23}; lane>=16: row=l-16, K {8..15, 24..31}
__device__ __forceinline__ void stage_A(bf16* As, int r, int half, uint4 v0, uint4 v1) {
    int mt = r >> 4, l = r & 15;
    // v0 = k-offsets +0..7 of this half -> lane l ; v1 = +8..15 -> lane l+16
    *(uint4*)(As + ((mt * 32 + l) << 4) + half * 8)        = v0;
    *(uint4*)(As + ((mt * 32 + l + 16) << 4) + half * 8)   = v1;
}

// Swizzled B staging: thread holds 8 fp32 of weight row k (local bk), local cols bn..bn+7.
//   B frag layout: lane l<16: col=l, K {0..7, 16..23} ; lane>=16: col=l-16, K {8..15, 24..31}
__device__ __forceinline__ void stage_B(bf16* Bs, int bk, int bn, float4 b0, float4 b1) {
    int nt = bn >> 4;
    int s  = (bn & 15) + ((bk & 8) ? 16 : 0);
    bf16* db = Bs + ((nt * 32 + s) << 4) + ((bk & 7) + ((bk & 16) >> 1));
    db[0 * 16] = (bf16)b0.x; db[1 * 16] = (bf16)b0.y;
    db[2 * 16] = (bf16)b0.z; db[3 * 16] = (bf16)b0.w;
    db[4 * 16] = (bf16)b1.x; db[5 * 16] = (bf16)b1.y;
    db[6 * 16] = (bf16)b1.z; db[7 * 16] = (bf16)b1.w;
}

// ---------------------------------------------------------------------------
// fp32 -> bf16 activation conversion
// ---------------------------------------------------------------------------
__global__ void convert_x_kernel(const float* __restrict__ x, bf16* __restrict__ xb) {
    long i = ((long)blockIdx.x * blockDim.x + threadIdx.x) * 8;
    const float4* s = (const float4*)(x + i);
    float4 a = s[0], b = s[1];
    union { bf16 h[8]; uint4 u; } o;
    o.h[0] = (bf16)a.x; o.h[1] = (bf16)a.y; o.h[2] = (bf16)a.z; o.h[3] = (bf16)a.w;
    o.h[4] = (bf16)b.x; o.h[5] = (bf16)b.y; o.h[6] = (bf16)b.z; o.h[7] = (bf16)b.w;
    *((uint4*)(xb + i)) = o.u;
}

__global__ void zero_counts_kernel(int* __restrict__ counts) {
    if (threadIdx.x < E_) counts[threadIdx.x] = 0;
}

// ---------------------------------------------------------------------------
// Router: one wave per token. logits -> top-4 -> renormalized softmax weights,
// builds per-expert token lists.
// ---------------------------------------------------------------------------
__global__ void router_kernel(const float* __restrict__ x, const float* __restrict__ rw,
                              int* __restrict__ counts, int* __restrict__ list,
                              int* __restrict__ slot_e, int* __restrict__ slot_o,
                              float* __restrict__ cw) {
    int lane = threadIdx.x & 31;
    int t = blockIdx.x * (blockDim.x >> 5) + (threadIdx.x >> 5);
    if (t >= T_) return;
    const float* xt = x + (long)t * D_;

    float logits[E_];
#pragma unroll
    for (int e = 0; e < E_; ++e) {
        const float* we = rw + (long)e * D_;
        float s = 0.f;
        for (int d = lane; d < D_; d += 32) s += xt[d] * we[d];
#pragma unroll
        for (int m = 16; m >= 1; m >>= 1) s += __shfl_xor(s, m, 32);
        logits[e] = s;
    }
    int   idx[TOPK_];
    float val[TOPK_];
    bool used[E_];
#pragma unroll
    for (int e = 0; e < E_; ++e) used[e] = false;
#pragma unroll
    for (int k = 0; k < TOPK_; ++k) {
        float best = -3.4e38f; int bi = 0;
#pragma unroll
        for (int e = 0; e < E_; ++e)
            if (!used[e] && logits[e] > best) { best = logits[e]; bi = e; }
        used[bi] = true; idx[k] = bi; val[k] = best;
    }
    float w[TOPK_], den = 0.f;
#pragma unroll
    for (int k = 0; k < TOPK_; ++k) { w[k] = __expf(val[k] - val[0]); den += w[k]; }
#pragma unroll
    for (int k = 0; k < TOPK_; ++k) w[k] /= den;

    if (lane == 0) {
        for (int k = 0; k < TOPK_; ++k) {
            int e   = idx[k];
            int off = atomicAdd(&counts[e], 1);
            list[e * T_ + off]    = t;
            slot_e[t * TOPK_ + k] = e;
            slot_o[t * TOPK_ + k] = off;
            cw[t * TOPK_ + k]     = w[k];
        }
    }
}

__global__ void scan_kernel(const int* __restrict__ counts, int* __restrict__ basev) {
    if (threadIdx.x == 0) {
        int s = 0;
        for (int e = 0; e < E_; ++e) { basev[e] = s; s += counts[e]; }
        basev[E_] = s;
    }
}

// ---------------------------------------------------------------------------
// Gated GEMM:  H = silu(A @ Bg) * (A @ Bu)
// grid: (N/BN, ceil(Mmax/BM), E or 1), block 256 = 8 waves (4x2 of 32x32)
// ---------------------------------------------------------------------------
__global__ void __launch_bounds__(256)
gated_gemm_kernel(const bf16* __restrict__ A, const float* __restrict__ Bg,
                  const float* __restrict__ Bu, long strideB, int K, int N,
                  const int* __restrict__ counts, const int* __restrict__ basev,
                  const int* __restrict__ list, int Mfixed,
                  bf16* __restrict__ Hout) {
    int e = blockIdx.z;
    int M = counts ? counts[e] : Mfixed;
    int m0 = blockIdx.y * BM;
    if (m0 >= M) return;                       // block-uniform exit
    int n0 = blockIdx.x * BN;
    int rowbase = basev ? basev[e] : 0;
    const int* tl = list ? (list + (long)e * T_) : nullptr;
    const float* bgp = Bg + (long)e * strideB;
    const float* bup = Bu + (long)e * strideB;

    __shared__ bf16 As[BM * BK];    // 8 KB, swizzled [8][32][16]
    __shared__ bf16 Bgs[BK * BN];   // 4 KB, swizzled [4][32][16]
    __shared__ bf16 Bus[BK * BN];

    int tid = threadIdx.x;
    int lane = tid & 31, wave = tid >> 5;
    int wm = (wave >> 1) * 32;                 // wave row offset in block tile
    int wn = (wave & 1) * 32;                  // wave col offset
    int amt = wm >> 4;                         // A tile16 base
    int bnt = wn >> 4;                         // B tile16 base

    v8f z8 = {0.f, 0.f, 0.f, 0.f, 0.f, 0.f, 0.f, 0.f};
    v8f cg[2][2] = {{z8, z8}, {z8, z8}};
    v8f cu[2][2] = {{z8, z8}, {z8, z8}};

    // A staging: thread -> (row = tid/2, k-half = tid&1)
    int ar = tid >> 1;
    int ah = tid & 1;
    int arow = m0 + ar;
    const bf16* asrc = nullptr;
    if (arow < M) {
        int srcrow = tl ? tl[arow] : arow;
        asrc = A + (long)srcrow * K + ah * 16;
    }
    // B staging: thread -> (k = tid/8, 8 cols at (tid&7)*8)
    int bk = tid >> 3;
    int bn = (tid & 7) * 8;

    for (int k0 = 0; k0 < K; k0 += BK) {
        {   // A tile (zero-pad ragged rows)
            uint4 v0 = {0u,0u,0u,0u}, v1 = {0u,0u,0u,0u};
            if (asrc) { const uint4* s = (const uint4*)(asrc + k0); v0 = s[0]; v1 = s[1]; }
            stage_A(As, ar, ah, v0, v1);
        }
        {   // B tiles: fp32 -> bf16, swizzled
            const float4* sg = (const float4*)(bgp + (long)(k0 + bk) * N + (n0 + bn));
            const float4* su = (const float4*)(bup + (long)(k0 + bk) * N + (n0 + bn));
            float4 g0 = sg[0], g1 = sg[1], u0 = su[0], u1 = su[1];
            stage_B(Bgs, bk, bn, g0, g1);
            stage_B(Bus, bk, bn, u0, u1);
        }
        __syncthreads();

        v16bf a0  = frag_ld(As, amt);
        v16bf a1  = frag_ld(As, amt + 1);
        v16bf bg0 = frag_ld(Bgs, bnt);
        v16bf bg1 = frag_ld(Bgs, bnt + 1);
        v16bf bu0 = frag_ld(Bus, bnt);
        v16bf bu1 = frag_ld(Bus, bnt + 1);

        cg[0][0] = wmma_bf16(a0, bg0, cg[0][0]);
        cu[0][0] = wmma_bf16(a0, bu0, cu[0][0]);
        cg[0][1] = wmma_bf16(a0, bg1, cg[0][1]);
        cu[0][1] = wmma_bf16(a0, bu1, cu[0][1]);
        cg[1][0] = wmma_bf16(a1, bg0, cg[1][0]);
        cu[1][0] = wmma_bf16(a1, bu0, cu[1][0]);
        cg[1][1] = wmma_bf16(a1, bg1, cg[1][1]);
        cu[1][1] = wmma_bf16(a1, bu1, cu[1][1]);
        __syncthreads();
    }

    // C layout: VGPR j -> M = j (lanes 0-15) or j+8 (lanes 16-31); N = lane&15
    int mofs = (lane >> 4) * 8;
    int ncol = lane & 15;
#pragma unroll
    for (int mi = 0; mi < 2; ++mi)
#pragma unroll
        for (int ni = 0; ni < 2; ++ni)
#pragma unroll
            for (int j = 0; j < 8; ++j) {
                int m = m0 + wm + mi * 16 + mofs + j;
                if (m < M) {
                    float g = cg[mi][ni][j];
                    float u = cu[mi][ni][j];
                    float h = (g / (1.0f + __expf(-g))) * u;   // silu(g)*u
                    int n = n0 + wn + ni * 16 + ncol;
                    Hout[(long)(rowbase + m) * N + n] = (bf16)h;
                }
            }
}

// ---------------------------------------------------------------------------
// Plain GEMM:  out = A @ B   (A bf16 compacted rows, B fp32->bf16)
// out: bf16 (expert down) or fp32 (shared down, direct to d_out)
// ---------------------------------------------------------------------------
__global__ void __launch_bounds__(256)
plain_gemm_kernel(const bf16* __restrict__ A, const float* __restrict__ B,
                  long strideB, int K, int N,
                  const int* __restrict__ counts, const int* __restrict__ basev,
                  int Mfixed, bf16* __restrict__ outb, float* __restrict__ outf) {
    int e = blockIdx.z;
    int M = counts ? counts[e] : Mfixed;
    int m0 = blockIdx.y * BM;
    if (m0 >= M) return;
    int n0 = blockIdx.x * BN;
    int rowbase = basev ? basev[e] : 0;
    const float* bp = B + (long)e * strideB;

    __shared__ bf16 As[BM * BK];
    __shared__ bf16 Bs[BK * BN];

    int tid = threadIdx.x;
    int lane = tid & 31, wave = tid >> 5;
    int wm = (wave >> 1) * 32;
    int wn = (wave & 1) * 32;
    int amt = wm >> 4;
    int bnt = wn >> 4;

    v8f z8 = {0.f, 0.f, 0.f, 0.f, 0.f, 0.f, 0.f, 0.f};
    v8f c[2][2] = {{z8, z8}, {z8, z8}};

    int ar = tid >> 1;
    int ah = tid & 1;
    int arow = m0 + ar;
    const bf16* asrc = (arow < M) ? (A + (long)(rowbase + arow) * K + ah * 16) : nullptr;
    int bk = tid >> 3;
    int bn = (tid & 7) * 8;

    for (int k0 = 0; k0 < K; k0 += BK) {
        {
            uint4 v0 = {0u,0u,0u,0u}, v1 = {0u,0u,0u,0u};
            if (asrc) { const uint4* s = (const uint4*)(asrc + k0); v0 = s[0]; v1 = s[1]; }
            stage_A(As, ar, ah, v0, v1);
        }
        {
            const float4* sb = (const float4*)(bp + (long)(k0 + bk) * N + (n0 + bn));
            float4 b0 = sb[0], b1 = sb[1];
            stage_B(Bs, bk, bn, b0, b1);
        }
        __syncthreads();

        v16bf a0 = frag_ld(As, amt);
        v16bf a1 = frag_ld(As, amt + 1);
        v16bf b0 = frag_ld(Bs, bnt);
        v16bf b1 = frag_ld(Bs, bnt + 1);
        c[0][0] = wmma_bf16(a0, b0, c[0][0]);
        c[0][1] = wmma_bf16(a0, b1, c[0][1]);
        c[1][0] = wmma_bf16(a1, b0, c[1][0]);
        c[1][1] = wmma_bf16(a1, b1, c[1][1]);
        __syncthreads();
    }

    int mofs = (lane >> 4) * 8;
    int ncol = lane & 15;
#pragma unroll
    for (int mi = 0; mi < 2; ++mi)
#pragma unroll
        for (int ni = 0; ni < 2; ++ni)
#pragma unroll
            for (int j = 0; j < 8; ++j) {
                int m = m0 + wm + mi * 16 + mofs + j;
                if (m < M) {
                    int n = n0 + wn + ni * 16 + ncol;
                    float v = c[mi][ni][j];
                    if (outf) outf[(long)m * N + n] = v;
                    else      outb[(long)(rowbase + m) * N + n] = (bf16)v;
                }
            }
}

// ---------------------------------------------------------------------------
// Combine: out[t,d] (holds shared-expert result) += sum_k w_k * y[slot,d]
// ---------------------------------------------------------------------------
__global__ void combine_kernel(float* __restrict__ out, const bf16* __restrict__ y,
                               const int* __restrict__ slot_e, const int* __restrict__ slot_o,
                               const float* __restrict__ cw, const int* __restrict__ basev) {
    int t = blockIdx.y;
    int d = blockIdx.x * blockDim.x + threadIdx.x;
    float acc = out[(long)t * D_ + d];
#pragma unroll
    for (int k = 0; k < TOPK_; ++k) {
        int e   = slot_e[t * TOPK_ + k];
        int pos = basev[e] + slot_o[t * TOPK_ + k];
        acc += cw[t * TOPK_ + k] * (float)y[(long)pos * D_ + d];
    }
    out[(long)t * D_ + d] = acc;
}

// ---------------------------------------------------------------------------
extern "C" void kernel_launch(void* const* d_in, const int* in_sizes, int n_in,
                              void* d_out, int out_size, void* d_ws, size_t ws_size,
                              hipStream_t stream) {
    const float* x   = (const float*)d_in[0];
    const float* rw  = (const float*)d_in[1];
    const float* wg  = (const float*)d_in[2];
    const float* wu  = (const float*)d_in[3];
    const float* wd  = (const float*)d_in[4];
    const float* swg = (const float*)d_in[5];
    const float* swu = (const float*)d_in[6];
    const float* swd = (const float*)d_in[7];
    float* out = (float*)d_out;

    // workspace layout
    char* p = (char*)d_ws;
    bf16* xb = (bf16*)p;   p += (size_t)T_ * D_ * 2;                 // 10.5 MB
    bf16* h  = (bf16*)p;   p += (size_t)T_ * TOPK_ * F_ * 2;         // 27.3 MB
    bf16* yv = (bf16*)p;   p += (size_t)T_ * TOPK_ * D_ * 2;         // 41.9 MB
    bf16* hs = (bf16*)p;   p += (size_t)T_ * FS_ * 2;                // 13.6 MB
    int* counts = (int*)p; p += 64;
    int* basev  = (int*)p; p += 128;
    int* list   = (int*)p; p += (size_t)E_ * T_ * 4;                 // 131 KB
    int* slot_e = (int*)p; p += (size_t)T_ * TOPK_ * 4;
    int* slot_o = (int*)p; p += (size_t)T_ * TOPK_ * 4;
    float* cw   = (float*)p; p += (size_t)T_ * TOPK_ * 4;

    convert_x_kernel<<<(T_ * D_) / (256 * 8), 256, 0, stream>>>(x, xb);
    zero_counts_kernel<<<1, 64, 0, stream>>>(counts);
    router_kernel<<<T_ / 8, 256, 0, stream>>>(x, rw, counts, list, slot_e, slot_o, cw);
    scan_kernel<<<1, 32, 0, stream>>>(counts, basev);

    // Expert gate/up + SiLU -> h (compacted rows)
    gated_gemm_kernel<<<dim3(F_ / BN, T_ / BM, E_), 256, 0, stream>>>(
        xb, wg, wu, (long)D_ * F_, D_, F_, counts, basev, list, 0, h);
    // Shared gate/up + SiLU -> hs
    gated_gemm_kernel<<<dim3(FS_ / BN, T_ / BM, 1), 256, 0, stream>>>(
        xb, swg, swu, 0, D_, FS_, nullptr, nullptr, nullptr, T_, hs);
    // Expert down: y (bf16, compacted rows)
    plain_gemm_kernel<<<dim3(D_ / BN, T_ / BM, E_), 256, 0, stream>>>(
        h, wd, (long)F_ * D_, F_, D_, counts, basev, 0, yv, nullptr);
    // Shared down: fp32 straight into d_out (every element written)
    plain_gemm_kernel<<<dim3(D_ / BN, T_ / BM, 1), 256, 0, stream>>>(
        hs, swd, 0, FS_, D_, nullptr, nullptr, T_, nullptr, out);
    // Weighted combine of top-4 expert outputs
    combine_kernel<<<dim3(D_ / 256, T_), 256, 0, stream>>>(out, yv, slot_e, slot_o, cw, basev);
}
